// LlamaDiffSparseKVAttention_78082505441329
// MI455X (gfx1250) — matrix operand order repaired and checked by reading, not compiled, not run on hardware
//
#include <hip/hip_runtime.h>

// ---------------- problem constants ----------------
#define S_     2048
#define HID_   4096
#define HQ_    32
#define HKV_   8
#define D_     128
#define G_     4
#define OBS_   128
#define W_     32
#define SINK_  2
#define KKEEP_ 39          // ceil((1-0.7)*128)
#define SCALE_ 0.08838834764831845f  // 1/sqrt(128)

typedef __attribute__((ext_vector_type(16))) __bf16 v16bf;
typedef __attribute__((ext_vector_type(8)))  float  v8f;
typedef __attribute__((ext_vector_type(4)))  int    v4i_;

static __device__ __forceinline__ v8f wmma_bf16(v16bf a, v16bf b, v8f c) {
  return __builtin_amdgcn_wmma_f32_16x16x32_bf16(false, a, false, b, (short)0, c, false, false);
}

// ---- optional CDNA5 async global->LDS copy (ASYNCcnt path), guarded ----
#if __has_builtin(__builtin_amdgcn_global_load_async_to_lds_b128) && __has_builtin(__builtin_amdgcn_s_wait_asynccnt)
#define ASYNC_LDS 1
static __device__ __forceinline__ void async_cp16(void* lds, const void* gmem) {
  __builtin_amdgcn_global_load_async_to_lds_b128(
      (__attribute__((address_space(1))) v4i_*)(v4i_*)const_cast<void*>(gmem),
      (__attribute__((address_space(3))) v4i_*)(v4i_*)lds,
      0, 0);
}
static __device__ __forceinline__ void async_wait0() {
  __builtin_amdgcn_s_wait_asynccnt(0);
}
#endif

// ---- WMMA fragment loaders, vectorized as 2 x ds_load_b128 (wave32) ----
// A 16x32 bf16 from row-major [row][ld]: lane row=lane&15, chunks at kb(=8*hi) and kb+16.
static __device__ __forceinline__ v16bf frag_a(const __bf16* p, int row0, int k0, int ld) {
  const int lane = threadIdx.x & 31;
  const __bf16* r = p + (size_t)(row0 + (lane & 15)) * ld + k0 + ((lane >> 4) << 3);
  union { uint4 u[2]; v16bf v; } t;
  t.u[0] = *(const uint4*)(r);
  t.u[1] = *(const uint4*)(r + 16);
  return t.v;
}
// B 32x16 bf16 from N-major [n][ld] storage: lane col n=lane&15, chunks at kb(=8*hi) and kb+16.
static __device__ __forceinline__ v16bf frag_b_nmajor(const __bf16* p, int n0, int k0, int ld) {
  const int lane = threadIdx.x & 31;
  const __bf16* r = p + (size_t)(n0 + (lane & 15)) * ld + k0 + ((lane >> 4) << 3);
  union { uint4 u[2]; v16bf v; } t;
  t.u[0] = *(const uint4*)(r);
  t.u[1] = *(const uint4*)(r + 16);
  return t.v;
}

// ---------------- GEMM: C[M,N] = A[M,K] * W[K,N], fp32 in/out, bf16 WMMA ----------------
#define BM 128
#define BN 128
#define BK 64
#define LDA 72    // row stride (bf16) for As   -> 144B, 16B multiple
#define LDBT 72   // row stride (bf16) for Bst  -> 144B, 16B multiple

__global__ __launch_bounds__(256) void gemm_bf16(const float* __restrict__ A,
                                                 const float* __restrict__ Wt,
                                                 float* __restrict__ C,
                                                 int M, int N, int K) {
  __shared__ alignas(16) __bf16 As[BM * LDA];     // row-major [m][k]
  __shared__ alignas(16) __bf16 Bst[BN * LDBT];   // transposed [n][k]
  const int tid  = threadIdx.x;
  const int lane = tid & 31;
  const int wave = tid >> 5;
  const int wm   = wave >> 1;   // 0..3 : 32-row strip
  const int wn   = wave & 1;    // 0..1 : 64-col strip
  const int row0 = blockIdx.y * BM;
  const int col0 = blockIdx.x * BN;

  v8f acc[2][4];
  for (int i = 0; i < 2; ++i)
    for (int j = 0; j < 4; ++j)
      for (int e = 0; e < 8; ++e) acc[i][j][e] = 0.f;

  for (int k0 = 0; k0 < K; k0 += BK) {
    // ---- batch all global tile loads into registers first (loads stay in flight) ----
    float4 fa[8], fw[8];
#pragma unroll
    for (int i = 0; i < 8; ++i) {               // A tile 128x64 f32 = 2048 float4
      int e  = i * 256 + tid;
      int r  = e >> 4;                           // 16 float4 per row
      int c4 = (e & 15) * 4;
      fa[i] = *(const float4*)&A[(size_t)(row0 + r) * K + k0 + c4];
    }
#pragma unroll
    for (int i = 0; i < 8; ++i) {               // W tile 64x128 f32 = 2048 float4
      int e  = i * 256 + tid;
      int r  = e >> 5;                           // 32 float4 per row
      int c4 = (e & 31) * 4;
      fw[i] = *(const float4*)&Wt[(size_t)(k0 + r) * N + col0 + c4];
    }
    // ---- convert + store to LDS ----
#pragma unroll
    for (int i = 0; i < 8; ++i) {
      int e  = i * 256 + tid;
      int r  = e >> 4;
      int c4 = (e & 15) * 4;
      union { __bf16 h[4]; uint2 u; } t;
      t.h[0] = (__bf16)fa[i].x; t.h[1] = (__bf16)fa[i].y;
      t.h[2] = (__bf16)fa[i].z; t.h[3] = (__bf16)fa[i].w;
      *(uint2*)&As[r * LDA + c4] = t.u;
    }
#pragma unroll
    for (int i = 0; i < 8; ++i) {
      int e  = i * 256 + tid;
      int r  = e >> 5;
      int c4 = (e & 31) * 4;
      Bst[(c4 + 0) * LDBT + r] = (__bf16)fw[i].x;
      Bst[(c4 + 1) * LDBT + r] = (__bf16)fw[i].y;
      Bst[(c4 + 2) * LDBT + r] = (__bf16)fw[i].z;
      Bst[(c4 + 3) * LDBT + r] = (__bf16)fw[i].w;
    }
    // prefetch next K-step tiles while computing this one
    if (k0 + BK < K) {
      __builtin_prefetch(&A[(size_t)(row0 + (tid >> 1)) * K + k0 + BK + (tid & 1) * 32], 0, 3);
      __builtin_prefetch(&Wt[(size_t)(k0 + BK + (tid >> 3)) * N + col0 + (tid & 7) * 16], 0, 3);
    }
    __syncthreads();

#pragma unroll
    for (int ks = 0; ks < 2; ++ks) {
      v16bf af0 = frag_a(As, wm * 32,      ks * 32, LDA);
      v16bf af1 = frag_a(As, wm * 32 + 16, ks * 32, LDA);
      v16bf bfr[4];
#pragma unroll
      for (int n0 = 0; n0 < 4; ++n0)
        bfr[n0] = frag_b_nmajor(Bst, wn * 64 + n0 * 16, ks * 32, LDBT);
#pragma unroll
      for (int n0 = 0; n0 < 4; ++n0) {
        acc[0][n0] = wmma_bf16(af0, bfr[n0], acc[0][n0]);
        acc[1][n0] = wmma_bf16(af1, bfr[n0], acc[1][n0]);
      }
    }
    __syncthreads();
  }

  const int hi = (lane >> 4) * 8;
  const int cc = lane & 15;
  for (int m0 = 0; m0 < 2; ++m0)
    for (int n0 = 0; n0 < 4; ++n0) {
#pragma unroll
      for (int r = 0; r < 8; ++r) {
        int rr = row0 + wm * 32 + m0 * 16 + hi + r;
        int nc = col0 + wn * 64 + n0 * 16 + cc;
        C[(size_t)rr * N + nc] = acc[m0][n0][r];
      }
    }
}

// ---------------- RoPE ----------------
__global__ void rope_q_kernel(float* __restrict__ Q, __bf16* __restrict__ Qb) {
  int t = blockIdx.x * blockDim.x + threadIdx.x;
  if (t >= S_ * HQ_ * (D_ / 2)) return;
  int i = t & 63;
  int h = (t >> 6) & (HQ_ - 1);
  int s = t >> 11;
  float inv = __powf(500000.0f, -(float)i / 64.0f);
  float sn, cs;
  __sincosf((float)s * inv, &sn, &cs);
  size_t base = (size_t)s * (HQ_ * D_) + (size_t)h * D_;
  float x1 = Q[base + i], x2 = Q[base + i + 64];
  float o1 = x1 * cs - x2 * sn;
  float o2 = x2 * cs + x1 * sn;
  Q[base + i] = o1; Q[base + i + 64] = o2;
  size_t ob = ((size_t)h * S_ + s) * D_;
  Qb[ob + i] = (__bf16)o1; Qb[ob + i + 64] = (__bf16)o2;
}

__global__ void rope_k_kernel(float* __restrict__ Kf) {
  int t = blockIdx.x * blockDim.x + threadIdx.x;
  if (t >= S_ * HKV_ * (D_ / 2)) return;
  int i = t & 63;
  int h = (t >> 6) & (HKV_ - 1);
  int s = t >> 9;
  float inv = __powf(500000.0f, -(float)i / 64.0f);
  float sn, cs;
  __sincosf((float)s * inv, &sn, &cs);
  size_t base = (size_t)s * (HKV_ * D_) + (size_t)h * D_;
  float x1 = Kf[base + i], x2 = Kf[base + i + 64];
  Kf[base + i]      = x1 * cs - x2 * sn;
  Kf[base + i + 64] = x2 * cs + x1 * sn;
}

__global__ void zero_f32(float* __restrict__ p, int n) {
  int i = blockIdx.x * blockDim.x + threadIdx.x;
  if (i < n) p[i] = 0.f;
}

// ---------------- observation-window importance ----------------
__global__ __launch_bounds__(256) void obs_importance(const float* __restrict__ Qf,
                                                      const float* __restrict__ Kf,
                                                      float* __restrict__ imp) {
  const int hkv  = blockIdx.y;
  const int oq   = blockIdx.x;
  const int qpos = S_ - OBS_ + oq;
  const int tid  = threadIdx.x;
  __shared__ alignas(16) float qv[G_][D_];
  __shared__ float sc[G_][S_];
  __shared__ float red[256];
  __shared__ float gmax[G_], gsum[G_];

  for (int i = tid; i < G_ * D_; i += 256) {
    int g = i >> 7, d = i & (D_ - 1);
    qv[g][d] = Qf[(size_t)qpos * (HQ_ * D_) + (size_t)(hkv * G_ + g) * D_ + d];
  }
  __syncthreads();

  const int nk = qpos + 1;
  float lmax[G_] = {-1e30f, -1e30f, -1e30f, -1e30f};
  for (int k = tid; k < nk; k += 256) {
    const float4* kp4 = (const float4*)(Kf + (size_t)k * (HKV_ * D_) + (size_t)hkv * D_);
    float d0 = 0.f, d1 = 0.f, d2 = 0.f, d3 = 0.f;
#pragma unroll 8
    for (int d4 = 0; d4 < D_ / 4; ++d4) {
      float4 kvv = kp4[d4];
      const float4 q0 = ((const float4*)qv[0])[d4];
      const float4 q1 = ((const float4*)qv[1])[d4];
      const float4 q2 = ((const float4*)qv[2])[d4];
      const float4 q3 = ((const float4*)qv[3])[d4];
      d0 += q0.x * kvv.x + q0.y * kvv.y + q0.z * kvv.z + q0.w * kvv.w;
      d1 += q1.x * kvv.x + q1.y * kvv.y + q1.z * kvv.z + q1.w * kvv.w;
      d2 += q2.x * kvv.x + q2.y * kvv.y + q2.z * kvv.z + q2.w * kvv.w;
      d3 += q3.x * kvv.x + q3.y * kvv.y + q3.z * kvv.z + q3.w * kvv.w;
    }
    d0 *= SCALE_; d1 *= SCALE_; d2 *= SCALE_; d3 *= SCALE_;
    sc[0][k] = d0; sc[1][k] = d1; sc[2][k] = d2; sc[3][k] = d3;
    lmax[0] = fmaxf(lmax[0], d0); lmax[1] = fmaxf(lmax[1], d1);
    lmax[2] = fmaxf(lmax[2], d2); lmax[3] = fmaxf(lmax[3], d3);
  }
  for (int g = 0; g < G_; ++g) {
    red[tid] = lmax[g]; __syncthreads();
    for (int s = 128; s > 0; s >>= 1) { if (tid < s) red[tid] = fmaxf(red[tid], red[tid + s]); __syncthreads(); }
    if (tid == 0) gmax[g] = red[0];
    __syncthreads();
  }
  float lsum[G_] = {0.f, 0.f, 0.f, 0.f};
  for (int k = tid; k < nk; k += 256) {
    for (int g = 0; g < G_; ++g) {
      float e = __expf(sc[g][k] - gmax[g]);
      sc[g][k] = e;
      lsum[g] += e;
    }
  }
  for (int g = 0; g < G_; ++g) {
    red[tid] = lsum[g]; __syncthreads();
    for (int s = 128; s > 0; s >>= 1) { if (tid < s) red[tid] += red[tid + s]; __syncthreads(); }
    if (tid == 0) gsum[g] = red[0];
    __syncthreads();
  }
  for (int k = tid; k < nk; k += 256) {
    float cnt = (float)min(OBS_, S_ - k);
    float contrib = 0.25f * (sc[0][k] / gsum[0] + sc[1][k] / gsum[1] +
                             sc[2][k] / gsum[2] + sc[3][k] / gsum[3]) / cnt;
    atomicAdd(&imp[hkv * S_ + k], contrib);
  }
}

// ---------------- exact quantiles via bit-space binary search ----------------
__global__ __launch_bounds__(256) void quantile_kernel(const float* __restrict__ imp,
                                                       float* __restrict__ thr) {
  const int n   = HKV_ * (S_ - W_);
  const int tid = threadIdx.x;
  __shared__ unsigned cnt[256];
  __shared__ float vals[4];
  const float ph = 0.95f * (float)(n - 1);
  const float pl = 0.20f * (float)(n - 1);
  int ks[4];
  ks[0] = (int)floorf(ph); ks[1] = min(ks[0] + 1, n - 1);
  ks[2] = (int)floorf(pl); ks[3] = min(ks[2] + 1, n - 1);

  for (int qi = 0; qi < 4; ++qi) {
    unsigned lo = 0u, hiu = 0xffffffffu;
    const unsigned target = (unsigned)(ks[qi] + 1);
    while (lo < hiu) {
      unsigned mid = lo + ((hiu - lo) >> 1);
      unsigned c = 0;
      for (int i = tid; i < n; i += 256) {
        int hh = i / (S_ - W_), cx = i % (S_ - W_);
        if (__float_as_uint(imp[hh * S_ + cx]) <= mid) ++c;
      }
      cnt[tid] = c; __syncthreads();
      for (int s = 128; s > 0; s >>= 1) { if (tid < s) cnt[tid] += cnt[tid + s]; __syncthreads(); }
      unsigned tot = cnt[0]; __syncthreads();
      if (tot >= target) hiu = mid; else lo = mid + 1;
    }
    if (tid == 0) vals[qi] = __uint_as_float(lo);
    __syncthreads();
  }
  if (tid == 0) {
    float fh = ph - floorf(ph), fl = pl - floorf(pl);
    thr[0] = vals[0] + (vals[1] - vals[0]) * fh;   // t_high
    thr[1] = vals[2] + (vals[3] - vals[2]) * fl;   // t_low
  }
}

// ---------------- level assignment + per-vector top-K sparsify ----------------
__global__ __launch_bounds__(128) void sparsify_kernel(const float* __restrict__ Kf,
                                                       const float* __restrict__ Vf,
                                                       const float* __restrict__ imp,
                                                       const float* __restrict__ thr,
                                                       __bf16* __restrict__ Ksp,
                                                       __bf16* __restrict__ Vsp,
                                                       unsigned char* __restrict__ evict) {
  const int pos = blockIdx.x, hkv = blockIdx.y, d = threadIdx.x;
  __shared__ float a[D_];
  __shared__ float red[D_];
  float kv = Kf[(size_t)pos * (HKV_ * D_) + (size_t)hkv * D_ + d];
  float vv = Vf[(size_t)pos * (HKV_ * D_) + (size_t)hkv * D_ + d];
  int level;
  if (pos >= S_ - W_ || pos < SINK_) level = 0;
  else {
    float im = imp[hkv * S_ + pos];
    level = (im >= thr[0]) ? 0 : ((im < thr[1]) ? 2 : 1);
  }
  if (d == 0) evict[hkv * S_ + pos] = (level == 2) ? 1 : 0;
  float ko = kv, vo = vv;
  if (level == 2) { ko = 0.f; vo = 0.f; }
  else if (level == 1) {
    // thr_k = max{ |k_i| : #{j : |k_j| >= |k_i|} >= KKEEP }  == sorted_abs[D-KKEEP]
    a[d] = fabsf(kv); __syncthreads();
    int cge = 0; float mya = a[d];
    for (int j = 0; j < D_; ++j) if (a[j] >= mya) ++cge;
    red[d] = (cge >= KKEEP_) ? mya : -1.f; __syncthreads();
    for (int s = 64; s > 0; s >>= 1) { if (d < s) red[d] = fmaxf(red[d], red[d + s]); __syncthreads(); }
    float tk = red[0]; __syncthreads();
    if (fabsf(kv) < tk) ko = 0.f;

    a[d] = fabsf(vv); __syncthreads();
    cge = 0; mya = a[d];
    for (int j = 0; j < D_; ++j) if (a[j] >= mya) ++cge;
    red[d] = (cge >= KKEEP_) ? mya : -1.f; __syncthreads();
    for (int s = 64; s > 0; s >>= 1) { if (d < s) red[d] = fmaxf(red[d], red[d + s]); __syncthreads(); }
    float tv = red[0];
    if (fabsf(vv) < tv) vo = 0.f;
  }
  size_t o = ((size_t)hkv * S_ + pos) * D_ + d;
  Ksp[o] = (__bf16)ko;
  Vsp[o] = (__bf16)vo;
}

// ---------------- flash attention (bf16 WMMA, online softmax) ----------------
#define LDK 136   // Ks row stride (bf16): 272B, 16B multiple
#define LDV 72    // Vst row stride (bf16): 144B, 16B multiple
#define LDP 72    // Ps row stride (bf16): 144B, 16B multiple

__global__ __launch_bounds__(128) void flash_attn(const __bf16* __restrict__ Qb,  // [HQ][S][D]
                                                  const __bf16* __restrict__ Kb,  // [HKV][S][D]
                                                  const __bf16* __restrict__ Vb,  // [HKV][S][D]
                                                  const unsigned char* __restrict__ evict, // [HKV][S]
                                                  float* __restrict__ Out) {      // [S][HQ*D]
  const int h    = blockIdx.y;
  const int hkv  = h >> 2;
  const int qblk = blockIdx.x;            // 64 q rows per block
  const int tid  = threadIdx.x;
  const int lane = tid & 31;
  const int wave = tid >> 5;              // 4 waves, 16 q rows each
  const int hi   = (lane >> 4) * 8;
  const int cc   = lane & 15;

  __shared__ alignas(16) __bf16 Ks[64 * LDK];        // key-major [key][d] (also Q staging)
  __shared__ alignas(16) __bf16 Vst[D_ * LDV];       // transposed [d][key]
  __shared__ alignas(16) __bf16 Ps[4][16 * LDP];     // per-wave P tile [qrow][key]

  // ---- stage Q tile (64x128): async global->LDS if available ----
  const __bf16* qptr = Qb + ((size_t)h * S_ + (size_t)qblk * 64) * D_;
  {
    const uint4* qp4 = (const uint4*)qptr;
#ifdef ASYNC_LDS
#pragma unroll
    for (int i = 0; i < 8; ++i) {
      int e = i * 128 + tid;
      int key = e >> 4, d8 = (e & 15) * 8;
      async_cp16(&Ks[key * LDK + d8], qp4 + e);
    }
    async_wait0();
#else
#pragma unroll
    for (int i = 0; i < 8; ++i) {
      int e = i * 128 + tid;
      int key = e >> 4, d8 = (e & 15) * 8;
      *(uint4*)&Ks[key * LDK + d8] = qp4[e];
    }
#endif
  }
  __syncthreads();
  v16bf qf[4];
#pragma unroll
  for (int kk = 0; kk < 4; ++kk) qf[kk] = frag_a(Ks, wave * 16, kk * 32, LDK);
  __syncthreads();

  float mrow[8], lrow[8];
  v8f oacc[8];
  for (int r = 0; r < 8; ++r) { mrow[r] = -1e30f; lrow[r] = 0.f; }
  for (int n = 0; n < 8; ++n)
    for (int e = 0; e < 8; ++e) oacc[n][e] = 0.f;

  const int nblocks = qblk + 1;            // causal: key blocks 0..qblk
  for (int j = 0; j < nblocks; ++j) {
    const int kbase = j * 64;
    const __bf16* kptr = Kb + ((size_t)hkv * S_ + kbase) * D_;
    const __bf16* vptr = Vb + ((size_t)hkv * S_ + kbase) * D_;
    const uint4* kp4 = (const uint4*)kptr;
    const uint4* vp4 = (const uint4*)vptr;

    // K tile: pure copy -> async to LDS when available
#ifdef ASYNC_LDS
#pragma unroll
    for (int i = 0; i < 8; ++i) {
      int e = i * 128 + tid;
      int key = e >> 4, d8 = (e & 15) * 8;
      async_cp16(&Ks[key * LDK + d8], kp4 + e);
    }
#else
#pragma unroll
    for (int i = 0; i < 8; ++i) {
      int e = i * 128 + tid;
      int key = e >> 4, d8 = (e & 15) * 8;
      *(uint4*)&Ks[key * LDK + d8] = kp4[e];
    }
#endif
    // V tile: batch loads, then transpose-scatter to [d][key]
    union { uint4 u; __bf16 hh[8]; } tv[8];
#pragma unroll
    for (int i = 0; i < 8; ++i) tv[i].u = vp4[i * 128 + tid];
#pragma unroll
    for (int i = 0; i < 8; ++i) {
      int e = i * 128 + tid;
      int key = e >> 4, d8 = (e & 15) * 8;
#pragma unroll
      for (int jj = 0; jj < 8; ++jj) Vst[(d8 + jj) * LDV + key] = tv[i].hh[jj];
    }
    // prefetch next KV block while computing this one
    if (j + 1 < nblocks) {
      __builtin_prefetch(kptr + 64 * D_ + tid * 64, 0, 3);
      __builtin_prefetch(vptr + 64 * D_ + tid * 64, 0, 3);
    }
#ifdef ASYNC_LDS
    async_wait0();
#endif
    __syncthreads();

    // S = Q * K^T  (16 q rows x 64 keys per wave); K tile is N-major for K^T
    v8f sacc[4];
    for (int n0 = 0; n0 < 4; ++n0)
      for (int e = 0; e < 8; ++e) sacc[n0][e] = 0.f;
#pragma unroll
    for (int kk = 0; kk < 4; ++kk) {
      v16bf bfr[4];
#pragma unroll
      for (int n0 = 0; n0 < 4; ++n0)
        bfr[n0] = frag_b_nmajor(Ks, n0 * 16, kk * 32, LDK);
#pragma unroll
      for (int n0 = 0; n0 < 4; ++n0)
        sacc[n0] = wmma_bf16(qf[kk], bfr[n0], sacc[n0]);
    }
    // scale + causal/evicted mask
    for (int n0 = 0; n0 < 4; ++n0) {
      int colg = kbase + n0 * 16 + cc;
      bool ev = evict[hkv * S_ + colg] != 0;
      for (int r = 0; r < 8; ++r) {
        int rowg = qblk * 64 + wave * 16 + hi + r;
        float s = sacc[n0][r] * SCALE_;
        if (colg > rowg || ev) s = -1e30f;
        sacc[n0][r] = s;
      }
    }
    // online softmax (rows live in 16-lane halves of the wave)
    float alpha[8];
    for (int r = 0; r < 8; ++r) {
      float mx = fmaxf(fmaxf(sacc[0][r], sacc[1][r]), fmaxf(sacc[2][r], sacc[3][r]));
      for (int off = 8; off >= 1; off >>= 1) mx = fmaxf(mx, __shfl_xor(mx, off, 16));
      float mnew = fmaxf(mrow[r], mx);
      alpha[r] = __expf(mrow[r] - mnew);
      mrow[r] = mnew;
      float ls = 0.f;
      for (int n0 = 0; n0 < 4; ++n0) {
        float p = __expf(sacc[n0][r] - mnew);
        ls += p;
        Ps[wave][(hi + r) * LDP + n0 * 16 + cc] = (__bf16)p;
      }
      for (int off = 8; off >= 1; off >>= 1) ls += __shfl_xor(ls, off, 16);
      lrow[r] = lrow[r] * alpha[r] + ls;
    }
    for (int n = 0; n < 8; ++n)
      for (int r = 0; r < 8; ++r) oacc[n][r] *= alpha[r];

    // O += P * V ; V tile stored N-major [d][key]
#pragma unroll
    for (int kk = 0; kk < 2; ++kk) {
      v16bf pf = frag_a(&Ps[wave][0], 0, kk * 32, LDP);
#pragma unroll
      for (int n = 0; n < 8; ++n) {
        v16bf bf_ = frag_b_nmajor(Vst, n * 16, kk * 32, LDV);
        oacc[n] = wmma_bf16(pf, bf_, oacc[n]);
      }
    }
    __syncthreads();
  }

  for (int n = 0; n < 8; ++n)
    for (int r = 0; r < 8; ++r) {
      int rowg = qblk * 64 + wave * 16 + hi + r;
      float invl = (lrow[r] > 0.f) ? 1.f / lrow[r] : 0.f;
      Out[(size_t)rowg * (HQ_ * D_) + h * D_ + n * 16 + cc] = oacc[n][r] * invl;
    }
}

// ---------------- launcher ----------------
extern "C" void kernel_launch(void* const* d_in, const int* in_sizes, int n_in,
                              void* d_out, int out_size, void* d_ws, size_t ws_size,
                              hipStream_t stream) {
  (void)in_sizes; (void)n_in; (void)out_size; (void)ws_size;
  const float* hs = (const float*)d_in[0];
  const float* wq = (const float*)d_in[1];
  const float* wk = (const float*)d_in[2];
  const float* wv = (const float*)d_in[3];
  const float* wo = (const float*)d_in[4];
  float* out = (float*)d_out;

  char* w = (char*)d_ws;
  size_t off = 0;
  auto alloc = [&](size_t bytes) { void* p = w + off; off += (bytes + 255) & ~(size_t)255; return p; };

  float*  qf  = (float*)alloc((size_t)S_ * HQ_ * D_ * 4);
  float*  kf  = (float*)alloc((size_t)S_ * HKV_ * D_ * 4);
  float*  vf  = (float*)alloc((size_t)S_ * HKV_ * D_ * 4);
  float*  ao  = (float*)alloc((size_t)S_ * HQ_ * D_ * 4);
  __bf16* qb  = (__bf16*)alloc((size_t)HQ_ * S_ * D_ * 2);
  __bf16* ksp = (__bf16*)alloc((size_t)HKV_ * S_ * D_ * 2);
  __bf16* vsp = (__bf16*)alloc((size_t)HKV_ * S_ * D_ * 2);
  float*  imp = (float*)alloc((size_t)HKV_ * S_ * 4);
  float*  thr = (float*)alloc(2 * 4);
  unsigned char* ev = (unsigned char*)alloc((size_t)HKV_ * S_);

  // QKV projections (bf16 WMMA GEMMs)
  gemm_bf16<<<dim3((HQ_ * D_) / BN, S_ / BM), 256, 0, stream>>>(hs, wq, qf, S_, HQ_ * D_, HID_);
  gemm_bf16<<<dim3((HKV_ * D_) / BN, S_ / BM), 256, 0, stream>>>(hs, wk, kf, S_, HKV_ * D_, HID_);
  gemm_bf16<<<dim3((HKV_ * D_) / BN, S_ / BM), 256, 0, stream>>>(hs, wv, vf, S_, HKV_ * D_, HID_);

  // RoPE (q also transposed to [HQ][S][D] bf16)
  rope_q_kernel<<<(S_ * HQ_ * (D_ / 2)) / 256, 256, 0, stream>>>(qf, qb);
  rope_k_kernel<<<(S_ * HKV_ * (D_ / 2)) / 256, 256, 0, stream>>>(kf);

  // importance scores from observation window
  zero_f32<<<(HKV_ * S_ + 255) / 256, 256, 0, stream>>>(imp, HKV_ * S_);
  obs_importance<<<dim3(OBS_, HKV_), 256, 0, stream>>>(qf, kf, imp);

  // quantile thresholds
  quantile_kernel<<<1, 256, 0, stream>>>(imp, thr);

  // level assignment + top-K magnitude sparsification -> bf16 K/V + evicted flags
  sparsify_kernel<<<dim3(S_, HKV_), D_, 0, stream>>>(kf, vf, imp, thr, ksp, vsp, ev);

  // flash attention over sparsified KV
  flash_attn<<<dim3(S_ / 64, HQ_), 128, 0, stream>>>(qb, ksp, vsp, ev, ao);

  // output projection
  gemm_bf16<<<dim3(HID_ / BN, S_ / BM), 256, 0, stream>>>(ao, wo, out, S_, HID_, HID_);
}
